// PhysicsAwareEmbedding_68307159876021
// MI455X (gfx1250) — compile-verified, matrix operand order checked by reading, not run on hardware
//
#include <hip/hip_runtime.h>
#include <hip/hip_bf16.h>

// Problem constants (from reference)
#define N_NODES 100000
#define N_EDGES 1600000
#define DIM     128
#define NTILES  ((N_NODES + 15) / 16)   // 6250 row-tiles of 16 nodes

typedef __bf16 bf16;
typedef __attribute__((ext_vector_type(16))) __bf16 v16bf;
typedef __attribute__((ext_vector_type(8)))  float  v8f;

// ---------- helpers ----------

__device__ __forceinline__ bf16 f2bf(float f) {
  unsigned u = __builtin_bit_cast(unsigned, f);
  unsigned r = u + 0x7FFFu + ((u >> 16) & 1u);   // round-to-nearest-even
  unsigned short hs = (unsigned short)(r >> 16);
  return __builtin_bit_cast(bf16, hs);
}

__device__ __forceinline__ float gelu_exact(float x) {
  return 0.5f * x * (1.0f + erff(x * 0.70710678118654752440f));
}

__device__ __forceinline__ v8f wmma_bf16(v16bf a, v16bf b, v8f c) {
  // D = A(16x32 bf16) * B(32x16 bf16) + C(16x16 f32)
  return __builtin_amdgcn_wmma_f32_16x16x32_bf16(false, a, false, b, (short)0, c,
                                                 false, false);
}

// B fragment: prepacked 512 bf16 per fragment, 16 contiguous bf16 per lane (32B load)
__device__ __forceinline__ v16bf load_bfrag(const bf16* __restrict__ packed,
                                            int frag, int lane) {
  return *reinterpret_cast<const v16bf*>(packed + (size_t)frag * 512 + lane * 16);
}

// A fragment (16x32 bf16) from a row-major f32 row. k0 = ktile*32 + half*8.
// ISA layout: lane m = L%16 (both halves), VGPR j holds K pair
// kbase(j,half) = (j%4)*2 + (j/4)*16 + half*8
__device__ __forceinline__ v16bf a_from_f32row(const float* __restrict__ rowp, int k0) {
  v16bf a;
#pragma unroll
  for (int j = 0; j < 8; ++j) {
    int k = k0 + ((j & 3) << 1) + ((j >> 2) << 4);
    float2 v = *reinterpret_cast<const float2*>(rowp + k);
    a[2 * j]     = f2bf(v.x);
    a[2 * j + 1] = f2bf(v.y);
  }
  return a;
}

// A fragment from a bf16 LDS tile row (leading dim 128)
__device__ __forceinline__ v16bf a_from_ldsrow(const bf16* rowp, int k0) {
  v16bf a;
#pragma unroll
  for (int j = 0; j < 8; ++j) {
    int k = k0 + ((j & 3) << 1) + ((j >> 2) << 4);
    a[2 * j]     = rowp[k];
    a[2 * j + 1] = rowp[k + 1];
  }
  return a;
}

// ---------- weight prepack: f32 (Krows x 128, row-major) -> bf16 B-fragments ----------
// fragment f = ktile*8 + ntile; within fragment: lane L stores its 16 vector elems
// contiguously so the GEMM-side load is one 32-byte contiguous load per lane.
__global__ __launch_bounds__(256) void prepack_kernel(const float* __restrict__ W,
                                                      bf16* __restrict__ dst,
                                                      int Krows, int nfrags) {
  int idx = blockIdx.x * blockDim.x + threadIdx.x;
  if (idx >= nfrags * 32) return;
  int frag = idx >> 5, lane = idx & 31;
  int ktile = frag >> 3, ntile = frag & 7;
  int n = lane & 15, half = lane >> 4;
  int col = ntile * 16 + n;
  bf16* out = dst + (size_t)frag * 512 + lane * 16;
#pragma unroll
  for (int e = 0; e < 16; ++e) {
    int j = e >> 1;
    int k = ktile * 32 + half * 8 + ((j & 3) << 1) + ((j >> 2) << 4) + (e & 1);
    float v = (k < Krows) ? W[(size_t)k * DIM + col] : 0.0f;
    out[e] = f2bf(v);
  }
}

// ---------- lift: h = gelu(nf @ w1 + b1) @ w2 + b2 ----------
__global__ __launch_bounds__(256) void lift_kernel(
    const float* __restrict__ x, const float* __restrict__ gf,
    const bf16* __restrict__ pw1, const float* __restrict__ b1,
    const bf16* __restrict__ pw2, const float* __restrict__ b2,
    float* __restrict__ h) {
  __shared__ bf16 lds[8][16 * DIM];
  const int wave = threadIdx.x >> 5, lane = threadIdx.x & 31;
  int tile = blockIdx.x * 8 + wave;
  const bool valid = (tile < NTILES);
  if (!valid) tile = NTILES - 1;
  const int m = lane & 15, half = lane >> 4, nl = m;
  const int row = tile * 16 + m;

  // Build A (16x32): [node_feats(6) | gf(8) | zero-pad to 32]
  v16bf a;
#pragma unroll
  for (int j = 0; j < 8; ++j) {
#pragma unroll
    for (int e = 0; e < 2; ++e) {
      int k = half * 8 + ((j & 3) << 1) + ((j >> 2) << 4) + e;
      float v = 0.0f;
      if (k < 6)       v = x[(size_t)row * 9 + 3 + k];
      else if (k < 14) v = gf[k - 6];
      a[2 * j + e] = f2bf(v);
    }
  }

  v8f zero = {};
  v8f acc[8];
#pragma unroll
  for (int t = 0; t < 8; ++t)
    acc[t] = wmma_bf16(a, load_bfrag(pw1, t, lane), zero);

  // gelu(. + b1) -> LDS bf16 tile (C layout: lanes0-15 M=r, lanes16-31 M=r+8)
  bf16* tl = lds[wave];
#pragma unroll
  for (int t = 0; t < 8; ++t) {
    const float bb = b1[t * 16 + nl];
#pragma unroll
    for (int r = 0; r < 8; ++r)
      tl[(r + half * 8) * DIM + t * 16 + nl] = f2bf(gelu_exact(acc[t][r] + bb));
  }
  __syncthreads();

  v8f o[8] = {};
  const bf16* arow = tl + m * DIM;
#pragma unroll
  for (int kt = 0; kt < 4; ++kt) {
    v16bf am = a_from_ldsrow(arow, kt * 32 + half * 8);
#pragma unroll
    for (int t = 0; t < 8; ++t)
      o[t] = wmma_bf16(am, load_bfrag(pw2, kt * 8 + t, lane), o[t]);
  }
  if (valid) {
#pragma unroll
    for (int t = 0; t < 8; ++t) {
      const float bb = b2[t * 16 + nl];
#pragma unroll
      for (int r = 0; r < 8; ++r)
        h[(size_t)(tile * 16 + r + half * 8) * DIM + t * 16 + nl] = o[t][r] + bb;
    }
  }
}

// ---------- nei = h @ W + b ----------
__global__ __launch_bounds__(256) void gemm_h_kernel(
    const float* __restrict__ hin, const bf16* __restrict__ pw,
    const float* __restrict__ bias, float* __restrict__ outm) {
  const int wave = threadIdx.x >> 5, lane = threadIdx.x & 31;
  int tile = blockIdx.x * 8 + wave;
  const bool valid = (tile < NTILES);
  if (!valid) tile = NTILES - 1;
  const int m = lane & 15, half = lane >> 4, nl = m;
  const float* rowp = hin + (size_t)(tile * 16 + m) * DIM;

  v8f o[8] = {};
#pragma unroll
  for (int kt = 0; kt < 4; ++kt) {
    __builtin_prefetch(pw + (size_t)((kt + 1) & 3) * 8 * 512, 0, 1);
    v16bf am = a_from_f32row(rowp, kt * 32 + half * 8);
#pragma unroll
    for (int t = 0; t < 8; ++t)
      o[t] = wmma_bf16(am, load_bfrag(pw, kt * 8 + t, lane), o[t]);
  }
  if (valid) {
#pragma unroll
    for (int t = 0; t < 8; ++t) {
      const float bb = bias[t * 16 + nl];
#pragma unroll
      for (int r = 0; r < 8; ++r)
        outm[(size_t)(tile * 16 + r + half * 8) * DIM + t * 16 + nl] = o[t][r] + bb;
    }
  }
}

// ---------- zero ----------
__global__ void zero_kernel(float* __restrict__ p, int n4) {
  int i = blockIdx.x * blockDim.x + threadIdx.x;
  if (i < n4) *reinterpret_cast<float4*>(p + (size_t)i * 4) = make_float4(0, 0, 0, 0);
}

// ---------- edge scatter: aggr[row] += nei[col] * ev  (1 wave / edge) ----------
__global__ __launch_bounds__(256) void scatter_kernel(
    const float* __restrict__ nei, const int* __restrict__ edge_index,
    const float* __restrict__ ev, float* __restrict__ aggr) {
  const int lane = threadIdx.x & 31;
  const int e = blockIdx.x * 8 + (threadIdx.x >> 5);
  if (e >= N_EDGES) return;
  const int r = edge_index[e];            // destination row
  const int c = edge_index[N_EDGES + e];  // source col
  const float w = ev[e];
  const float4 v = *reinterpret_cast<const float4*>(nei + (size_t)c * DIM + lane * 4);
  float* dst = aggr + (size_t)r * DIM + lane * 4;
  atomicAdd(dst + 0, v.x * w);
  atomicAdd(dst + 1, v.y * w);
  atomicAdd(dst + 2, v.z * w);
  atomicAdd(dst + 3, v.w * w);
}

// ---------- GCN update: h += gelu([h@Wself+b | aggr] @ gw1 + gb1) @ gw2 + gb2 ----------
__global__ __launch_bounds__(256) void upd_kernel(
    float* __restrict__ h, const float* __restrict__ aggr,
    const bf16* __restrict__ pwself, const float* __restrict__ bself,
    const bf16* __restrict__ pgw1, const float* __restrict__ gb1,
    const bf16* __restrict__ pgw2, const float* __restrict__ gb2) {
  __shared__ bf16 lds[8][16 * DIM];
  const int wave = threadIdx.x >> 5, lane = threadIdx.x & 31;
  int tile = blockIdx.x * 8 + wave;
  const bool valid = (tile < NTILES);
  if (!valid) tile = NTILES - 1;
  const int m = lane & 15, half = lane >> 4, nl = m;
  const float* hrow = h + (size_t)(tile * 16 + m) * DIM;
  const float* grow = aggr + (size_t)(tile * 16 + m) * DIM;
  bf16* tl = lds[wave];
  const bf16* lrow = tl + m * DIM;

  // self_f = h @ Wself
  v8f sf[8] = {};
#pragma unroll
  for (int kt = 0; kt < 4; ++kt) {
    v16bf am = a_from_f32row(hrow, kt * 32 + half * 8);
#pragma unroll
    for (int t = 0; t < 8; ++t)
      sf[t] = wmma_bf16(am, load_bfrag(pwself, kt * 8 + t, lane), sf[t]);
  }
  // + bself -> LDS (bf16)
#pragma unroll
  for (int t = 0; t < 8; ++t) {
    const float bb = bself[t * 16 + nl];
#pragma unroll
    for (int r = 0; r < 8; ++r)
      tl[(r + half * 8) * DIM + t * 16 + nl] = f2bf(sf[t][r] + bb);
  }
  __syncthreads();

  // g1 = [self_f | aggr] @ gw1   (K = 256: ktiles 0-3 self from LDS, 4-7 aggr from global)
  v8f g1[8] = {};
#pragma unroll
  for (int kt = 0; kt < 4; ++kt) {
    v16bf am = a_from_ldsrow(lrow, kt * 32 + half * 8);
#pragma unroll
    for (int t = 0; t < 8; ++t)
      g1[t] = wmma_bf16(am, load_bfrag(pgw1, kt * 8 + t, lane), g1[t]);
  }
#pragma unroll
  for (int kt = 0; kt < 4; ++kt) {
    v16bf am = a_from_f32row(grow, kt * 32 + half * 8);
#pragma unroll
    for (int t = 0; t < 8; ++t)
      g1[t] = wmma_bf16(am, load_bfrag(pgw1, (4 + kt) * 8 + t, lane), g1[t]);
  }
  __syncthreads();

  // gelu(g1 + gb1) -> LDS
#pragma unroll
  for (int t = 0; t < 8; ++t) {
    const float bb = gb1[t * 16 + nl];
#pragma unroll
    for (int r = 0; r < 8; ++r)
      tl[(r + half * 8) * DIM + t * 16 + nl] = f2bf(gelu_exact(g1[t][r] + bb));
  }
  __syncthreads();

  // g2 = mid @ gw2 ; h += g2 + gb2
  v8f g2[8] = {};
#pragma unroll
  for (int kt = 0; kt < 4; ++kt) {
    v16bf am = a_from_ldsrow(lrow, kt * 32 + half * 8);
#pragma unroll
    for (int t = 0; t < 8; ++t)
      g2[t] = wmma_bf16(am, load_bfrag(pgw2, kt * 8 + t, lane), g2[t]);
  }
  if (valid) {
#pragma unroll
    for (int t = 0; t < 8; ++t) {
      const float bb = gb2[t * 16 + nl];
#pragma unroll
      for (int r = 0; r < 8; ++r) {
        size_t idx = (size_t)(tile * 16 + r + half * 8) * DIM + t * 16 + nl;
        h[idx] += g2[t][r] + bb;
      }
    }
  }
}

// ---------- FiLM: gamma|beta = gelu(gf @ fw1 + fb1) @ fw2 + fb2 ----------
__global__ __launch_bounds__(256) void film_kernel(
    const float* __restrict__ gf, const float* __restrict__ fw1,
    const float* __restrict__ fb1, const float* __restrict__ fw2,
    const float* __restrict__ fb2, float* __restrict__ gammabeta) {
  __shared__ float t[DIM];
  const int tid = threadIdx.x;
  if (tid < DIM) {
    float s = fb1[tid];
#pragma unroll
    for (int k = 0; k < 8; ++k) s += gf[k] * fw1[k * DIM + tid];
    t[tid] = gelu_exact(s);
  }
  __syncthreads();
  float s = fb2[tid];
  for (int j = 0; j < DIM; ++j) s += t[j] * fw2[(size_t)j * 256 + tid];
  gammabeta[tid] = s;
}

// ---------- layernorm + FiLM apply (1 wave / node) ----------
__global__ __launch_bounds__(256) void final_kernel(
    const float* __restrict__ h, const float* __restrict__ gb,
    const float* __restrict__ ln_g, const float* __restrict__ ln_b,
    float* __restrict__ out) {
  const int lane = threadIdx.x & 31;
  const int node = blockIdx.x * 8 + (threadIdx.x >> 5);
  if (node >= N_NODES) return;
  const float4 v = *reinterpret_cast<const float4*>(h + (size_t)node * DIM + lane * 4);
  float s = v.x + v.y + v.z + v.w;
#pragma unroll
  for (int off = 16; off > 0; off >>= 1) s += __shfl_xor(s, off, 32);
  const float mu = s * (1.0f / 128.0f);
  const float d0 = v.x - mu, d1 = v.y - mu, d2 = v.z - mu, d3 = v.w - mu;
  float q = d0 * d0 + d1 * d1 + d2 * d2 + d3 * d3;
#pragma unroll
  for (int off = 16; off > 0; off >>= 1) q += __shfl_xor(q, off, 32);
  const float inv = rsqrtf(q * (1.0f / 128.0f) + 1e-5f);
  const int c = lane * 4;
  float4 o;
  o.x = gb[c + 0] * (d0 * inv * ln_g[c + 0] + ln_b[c + 0]) + gb[DIM + c + 0];
  o.y = gb[c + 1] * (d1 * inv * ln_g[c + 1] + ln_b[c + 1]) + gb[DIM + c + 1];
  o.z = gb[c + 2] * (d2 * inv * ln_g[c + 2] + ln_b[c + 2]) + gb[DIM + c + 2];
  o.w = gb[c + 3] * (d3 * inv * ln_g[c + 3] + ln_b[c + 3]) + gb[DIM + c + 3];
  *reinterpret_cast<float4*>(out + (size_t)node * DIM + c) = o;
}

// ---------- host ----------
extern "C" void kernel_launch(void* const* d_in, const int* in_sizes, int n_in,
                              void* d_out, int out_size, void* d_ws, size_t ws_size,
                              hipStream_t stream) {
  const float* x     = (const float*)d_in[0];
  const int*   ei    = (const int*)d_in[1];
  const float* ev    = (const float*)d_in[2];
  const float* gf    = (const float*)d_in[3];
  const float* lw1   = (const float*)d_in[4];
  const float* lb1   = (const float*)d_in[5];
  const float* lw2   = (const float*)d_in[6];
  const float* lb2   = (const float*)d_in[7];
  const float* wself = (const float*)d_in[8];
  const float* bself = (const float*)d_in[9];
  const float* wnei  = (const float*)d_in[10];
  const float* bnei  = (const float*)d_in[11];
  const float* gw1   = (const float*)d_in[12];
  const float* gb1   = (const float*)d_in[13];
  const float* gw2   = (const float*)d_in[14];
  const float* gb2   = (const float*)d_in[15];
  const float* lng   = (const float*)d_in[16];
  const float* lnb   = (const float*)d_in[17];
  const float* fw1   = (const float*)d_in[18];
  const float* fb1   = (const float*)d_in[19];
  const float* fw2   = (const float*)d_in[20];
  const float* fb2   = (const float*)d_in[21];
  float* out = (float*)d_out;

  char* ws = (char*)d_ws;
  size_t off = 0;
  auto alloc = [&](size_t bytes) -> void* {
    off = (off + 255) & ~(size_t)255;
    void* p = ws + off;
    off += bytes;
    return p;
  };

  const size_t HB = (size_t)N_NODES * DIM * sizeof(float);
  float* h    = (float*)alloc(HB);
  float* nei  = (float*)alloc(HB);
  float* aggr = (float*)alloc(HB);
  float* gbta = (float*)alloc(256 * sizeof(float));

  // packed bf16 weights: nfrags * 512 bf16 each
  bf16* p_lw1 = (bf16*)alloc((size_t)8 * 512 * 2);        // K=14 -> 1 ktile
  bf16* p_lw2 = (bf16*)alloc((size_t)32 * 512 * 2);       // K=128
  bf16* p_wnei[2]  = {(bf16*)alloc((size_t)32 * 512 * 2), (bf16*)alloc((size_t)32 * 512 * 2)};
  bf16* p_wself[2] = {(bf16*)alloc((size_t)32 * 512 * 2), (bf16*)alloc((size_t)32 * 512 * 2)};
  bf16* p_gw1[2]   = {(bf16*)alloc((size_t)64 * 512 * 2), (bf16*)alloc((size_t)64 * 512 * 2)};
  bf16* p_gw2[2]   = {(bf16*)alloc((size_t)32 * 512 * 2), (bf16*)alloc((size_t)32 * 512 * 2)};

  auto pp = [&](const float* W, bf16* dst, int Krows, int nfrags) {
    prepack_kernel<<<(nfrags * 32 + 255) / 256, 256, 0, stream>>>(W, dst, Krows, nfrags);
  };
  pp(lw1, p_lw1, 14, 8);
  pp(lw2, p_lw2, 128, 32);
  for (int i = 0; i < 2; ++i) {
    pp(wnei  + (size_t)i * 128 * 128, p_wnei[i],  128, 32);
    pp(wself + (size_t)i * 128 * 128, p_wself[i], 128, 32);
    pp(gw1   + (size_t)i * 256 * 128, p_gw1[i],   256, 64);
    pp(gw2   + (size_t)i * 128 * 128, p_gw2[i],   128, 32);
  }

  const int tileBlocks = (NTILES + 7) / 8;   // 8 waves (tiles) per 256-thread block
  lift_kernel<<<tileBlocks, 256, 0, stream>>>(x, gf, p_lw1, lb1, p_lw2, lb2, h);
  film_kernel<<<1, 256, 0, stream>>>(gf, fw1, fb1, fw2, fb2, gbta);

  const int n4 = N_NODES * DIM / 4;
  for (int i = 0; i < 2; ++i) {
    gemm_h_kernel<<<tileBlocks, 256, 0, stream>>>(h, p_wnei[i], bnei + i * DIM, nei);
    zero_kernel<<<(n4 + 255) / 256, 256, 0, stream>>>(aggr, n4);
    scatter_kernel<<<(N_EDGES + 7) / 8, 256, 0, stream>>>(nei, ei, ev, aggr);
    upd_kernel<<<tileBlocks, 256, 0, stream>>>(h, aggr, p_wself[i], bself + i * DIM,
                                               p_gw1[i], gb1 + i * DIM,
                                               p_gw2[i], gb2 + i * DIM);
  }

  final_kernel<<<(N_NODES + 7) / 8, 256, 0, stream>>>(h, gbta, lng, lnb, out);
  (void)in_sizes; (void)n_in; (void)out_size; (void)ws_size;
}
